// RNN_17343077941673
// MI455X (gfx1250) — matrix operand order repaired and checked by reading, not compiled, not run on hardware
//
#include <hip/hip_runtime.h>
#include <hip/hip_bf16.h>
#include <stdint.h>

// Problem dims (compile-time for best codegen)
#define B_    256
#define NS    512
#define TT    512
#define HID   1024
#define EPSV  1e-5f

typedef __attribute__((ext_vector_type(16))) __bf16 v16bf;
typedef __attribute__((ext_vector_type(8)))  float  v8f;

struct __attribute__((aligned(16))) U4 { uint32_t a, b, c, d; };

union BFrag {
  v16bf v;
  U4    q[2];
};

// CDNA5 async global->LDS path (gfx1250). Guarded: if the toolchain lacks the
// builtins we fall back to global_load_b128 + ds_store_b128 staging.
#if defined(__has_builtin)
#  if __has_builtin(__builtin_amdgcn_global_load_async_to_lds_b128) && \
      __has_builtin(__builtin_amdgcn_s_wait_asynccnt)
#    define USE_ASYNC_LDS 1
#  endif
#endif
#ifndef USE_ASYNC_LDS
#  define USE_ASYNC_LDS 0
#endif

#if USE_ASYNC_LDS
typedef int v4i __attribute__((vector_size(16)));   // builtin's element type
#endif

__device__ __forceinline__ void async_cp16(const unsigned short* g,
                                           unsigned short* l) {
#if USE_ASYNC_LDS
  // vector async copy: each lane moves 16B global -> LDS, tracked by ASYNCcnt
  __builtin_amdgcn_global_load_async_to_lds_b128(
      (__attribute__((address_space(1))) v4i*)g,
      (__attribute__((address_space(3))) v4i*)l,
      0, 0);
#else
  *reinterpret_cast<U4*>(l) = *reinterpret_cast<const U4*>(g);
#endif
}

__device__ __forceinline__ void wait_async_4() {
#if USE_ASYNC_LDS
  __builtin_amdgcn_s_wait_asynccnt(4);   // newest 4 (next chunk) stay in flight
#endif
}
__device__ __forceinline__ void wait_async_0() {
#if USE_ASYNC_LDS
  __builtin_amdgcn_s_wait_asynccnt(0);
#endif
}

__device__ __forceinline__ unsigned short f2bf(float f) {
  union { float f; uint32_t u; } x; x.f = f;
  uint32_t r = x.u + 0x7fffu + ((x.u >> 16) & 1u);   // round-to-nearest-even
  return (unsigned short)(r >> 16);
}

// ---------------------------------------------------------------------------
// Elementwise helpers
// ---------------------------------------------------------------------------
__global__ void cvt_bf16_kernel(const float* __restrict__ s,
                                unsigned short* __restrict__ d, int n) {
  int i = blockIdx.x * blockDim.x + threadIdx.x;
  if (i < n) d[i] = f2bf(s[i]);
}

// Wh [HID, 2*HID] -> Whx [HID, HID] (cols 0..H-1), Whh [HID, HID] (cols H..2H-1)
__global__ void split_wh_kernel(const float* __restrict__ wh,
                                unsigned short* __restrict__ whx,
                                unsigned short* __restrict__ whh) {
  int i = blockIdx.x * blockDim.x + threadIdx.x;      // over HID*2*HID
  if (i >= HID * 2 * HID) return;
  int row = i >> 11;              // / (2*HID)
  int col = i & (2 * HID - 1);
  unsigned short v = f2bf(wh[i]);
  if (col < HID) whx[row * HID + col] = v;
  else           whh[row * HID + (col - HID)] = v;
}

__global__ void zero_bf_kernel(unsigned short* __restrict__ p, int n) {
  int i = blockIdx.x * blockDim.x + threadIdx.x;
  if (i < n) p[i] = 0;
}

// ---------------------------------------------------------------------------
// x [B, NS, TT] f32  ->  xt [TT, B, NS] bf16   (LDS-tiled 32x32 transpose)
// grid (TT/32, NS/32, B), block (32, 8)
// ---------------------------------------------------------------------------
__global__ __launch_bounds__(256) void transpose_x_kernel(
    const float* __restrict__ x, unsigned short* __restrict__ xt) {
  __shared__ float tile[32][33];
  int b  = blockIdx.z;
  int t0 = blockIdx.x * 32;
  int n0 = blockIdx.y * 32;
  int tx = threadIdx.x, ty = threadIdx.y;
#pragma unroll
  for (int i = 0; i < 4; ++i) {
    int n = n0 + ty + 8 * i;
    tile[ty + 8 * i][tx] = x[((size_t)b * NS + n) * TT + t0 + tx];
  }
  __syncthreads();
#pragma unroll
  for (int i = 0; i < 4; ++i) {
    int t = t0 + ty + 8 * i;
    xt[((size_t)t * B_ + b) * NS + n0 + tx] = f2bf(tile[tx][ty + 8 * i]);
  }
}

// ---------------------------------------------------------------------------
// Big GEMM:  out[M,N] = A[M,K](bf16) * W[N,K](bf16)^T + bias[N]
// Block tile 128x128 (8 waves, wave tile 64x32), K-chunk 32.
// A/B tiles double-buffered in LDS via CDNA5 async global->LDS copies;
// fragment reads via ds_load_b128. LDS rows padded to 40 elems (80B) so the
// 16 fragment rows map to distinct bank groups.
// grid (N/128, M/128), block 256.
// ---------------------------------------------------------------------------
#define LDS_STRIDE 40   // padded row stride in bf16 elements (80 bytes)

template <bool OUT_F32, bool OUT_BF16>
__global__ __launch_bounds__(256) void gemm_big_kernel(
    const unsigned short* __restrict__ A,   // [M,K]
    const unsigned short* __restrict__ W,   // [N,K]
    const float* __restrict__ bias,         // [N]
    float* __restrict__ outF,               // [M,N] (if OUT_F32)
    unsigned short* __restrict__ outB,      // [M,N] (if OUT_BF16)
    int M, int N, int K) {
  __shared__ unsigned short sA[2][128 * LDS_STRIDE];   // 2 x 10240 B
  __shared__ unsigned short sB[2][128 * LDS_STRIDE];   // 2 x 10240 B

  const int tid  = threadIdx.x;
  const int lane = tid & 31;
  const int wid  = tid >> 5;
  const int l15  = lane & 15;
  const int hl   = lane >> 4;                 // lane half
  const int mblk = blockIdx.y * 128;
  const int nblk = blockIdx.x * 128;
  const int wm   = (wid & 1) * 64;            // wave M offset within block
  const int wn   = (wid >> 1) * 32;           // wave N offset within block

  // Staging map: 128 rows x 32 elems = 512 x 16B chunks per tile; each of the
  // 256 threads copies 2 chunks of A and 2 of B per K-chunk.
  const int r0 = tid >> 2;                    // rows 0..63
  const int r1 = r0 + 64;                     // rows 64..127
  const int c0 = (tid & 3) * 8;               // element offset 0/8/16/24

  v8f acc[4][2] = {};

  auto stage = [&](int buf, int kc) {
    async_cp16(A + (size_t)(mblk + r0) * K + kc + c0, &sA[buf][r0 * LDS_STRIDE + c0]);
    async_cp16(A + (size_t)(mblk + r1) * K + kc + c0, &sA[buf][r1 * LDS_STRIDE + c0]);
    async_cp16(W + (size_t)(nblk + r0) * K + kc + c0, &sB[buf][r0 * LDS_STRIDE + c0]);
    async_cp16(W + (size_t)(nblk + r1) * K + kc + c0, &sB[buf][r1 * LDS_STRIDE + c0]);
  };

  stage(0, 0);
  int buf = 0;
  for (int kc = 0; kc < K; kc += 32, buf ^= 1) {
    if (kc + 32 < K) {
      stage(buf ^ 1, kc + 32);   // overlap next chunk's copy with this compute
      wait_async_4();            // wait for current buffer only
    } else {
      wait_async_0();
    }
    __syncthreads();             // all waves' staging for 'buf' visible

    BFrag a[4], b[2];
#pragma unroll
    for (int i = 0; i < 4; ++i) {
      const unsigned short* p = &sA[buf][(wm + 16 * i + l15) * LDS_STRIDE + hl * 8];
      a[i].q[0] = *reinterpret_cast<const U4*>(p);        // K = kb+0..7
      a[i].q[1] = *reinterpret_cast<const U4*>(p + 16);   // K = kb+16..23
    }
#pragma unroll
    for (int j = 0; j < 2; ++j) {
      const unsigned short* p = &sB[buf][(wn + 16 * j + l15) * LDS_STRIDE + hl * 8];
      b[j].q[0] = *reinterpret_cast<const U4*>(p);        // K = kb+0..7
      b[j].q[1] = *reinterpret_cast<const U4*>(p + 16);   // K = kb+16..23
    }
#pragma unroll
    for (int i = 0; i < 4; ++i)
#pragma unroll
      for (int j = 0; j < 2; ++j)
        acc[i][j] = __builtin_amdgcn_wmma_f32_16x16x32_bf16(
            false, a[i].v, false, b[j].v, (short)0, acc[i][j], false, false);

    __syncthreads();             // done reading 'buf' before it is re-staged
  }

#pragma unroll
  for (int j = 0; j < 2; ++j) {
    const int n = nblk + wn + 16 * j + l15;
    const float bn = bias ? bias[n] : 0.0f;
#pragma unroll
    for (int i = 0; i < 4; ++i) {
#pragma unroll
      for (int r = 0; r < 8; ++r) {
        const int m = mblk + wm + 16 * i + r + 8 * hl;   // C/D: VGPR r + lane half
        const float v = acc[i][j][r] + bn;
        if (OUT_F32)  outF[(size_t)m * N + n] = v;
        if (OUT_BF16) outB[(size_t)m * N + n] = f2bf(v);
      }
    }
  }
}

// ---------------------------------------------------------------------------
// Recurrence step: hOut = tanh(Gt + hIn @ Whh^T)     [B_, HID]
// One 16x16 WMMA tile per wave; 1024 tiles -> 128 blocks of 8 waves.
// Whh (2MB bf16) is resident in L2 across all 512 steps.
// ---------------------------------------------------------------------------
__global__ __launch_bounds__(256) void rnn_step_kernel(
    const float* __restrict__ Gt,            // [B_,HID] (bh already folded in)
    const unsigned short* __restrict__ hIn,  // [B_,HID] bf16
    const unsigned short* __restrict__ Whh,  // [HID,HID] bf16
    unsigned short* __restrict__ hOutB,      // [B_,HID] bf16
    float* __restrict__ hOutF) {             // [B_,HID] f32
  const int tid  = threadIdx.x;
  const int lane = tid & 31;
  const int wid  = tid >> 5;
  const int l15  = lane & 15;
  const int hl   = lane >> 4;
  const int tile = blockIdx.x * 8 + wid;      // 0..1023
  const int m0 = (tile & 15) * 16;
  const int n0 = (tile >> 4) * 16;

  v8f acc = {};
  const unsigned short* pA = hIn + (size_t)(m0 + l15) * HID + hl * 8;
  const unsigned short* pB = Whh + (size_t)(n0 + l15) * HID + hl * 8;

  for (int kc = 0; kc < HID; kc += 32) {
    BFrag a, b;
    a.q[0] = *reinterpret_cast<const U4*>(pA + kc);
    a.q[1] = *reinterpret_cast<const U4*>(pA + kc + 16);
    b.q[0] = *reinterpret_cast<const U4*>(pB + kc);
    b.q[1] = *reinterpret_cast<const U4*>(pB + kc + 16);
    if (kc + 32 < HID) {
      __builtin_prefetch(pA + kc + 32, 0, 3);
      __builtin_prefetch(pB + kc + 32, 0, 3);
    }
    acc = __builtin_amdgcn_wmma_f32_16x16x32_bf16(
        false, a.v, false, b.v, (short)0, acc, false, false);
  }

  const int n = n0 + l15;
#pragma unroll
  for (int r = 0; r < 8; ++r) {
    const int m = m0 + r + 8 * hl;
    const float v = tanhf(Gt[(size_t)m * HID + n] + acc[r]);
    hOutF[(size_t)m * HID + n] = v;
    hOutB[(size_t)m * HID + n] = f2bf(v);
  }
}

// ---------------------------------------------------------------------------
// BatchNorm1d batch stats (biased var), per hidden unit j over B_ rows.
// ---------------------------------------------------------------------------
__global__ void bn_stats_kernel(const float* __restrict__ h,
                                float* __restrict__ mean,
                                float* __restrict__ rstd) {
  int j = blockIdx.x * blockDim.x + threadIdx.x;
  if (j >= HID) return;
  float s = 0.0f, ss = 0.0f;
  for (int b = 0; b < B_; ++b) {
    float v = h[(size_t)b * HID + j];
    s += v; ss += v * v;
  }
  float m = s * (1.0f / B_);
  float var = ss * (1.0f / B_) - m * m;
  mean[j] = m;
  rstd[j] = rsqrtf(var + EPSV);
}

__global__ void bn_apply_kernel(const float* __restrict__ h,
                                const float* __restrict__ mean,
                                const float* __restrict__ rstd,
                                const float* __restrict__ gamma,
                                const float* __restrict__ beta,
                                unsigned short* __restrict__ hn) {
  int i = blockIdx.x * blockDim.x + threadIdx.x;   // over B_*HID
  if (i >= B_ * HID) return;
  int j = i & (HID - 1);
  float v = gamma[j] * (h[i] - mean[j]) * rstd[j] + beta[j];
  hn[i] = f2bf(v);
}

// ---------------------------------------------------------------------------
// Final layer: out[B_,NS] = hn[B_,HID](bf16) @ Wl[NS,HID](bf16)^T + bl
// 16x16 WMMA tile per wave; 512 tiles -> 64 blocks.
// ---------------------------------------------------------------------------
__global__ __launch_bounds__(256) void gemm_out_kernel(
    const unsigned short* __restrict__ A,    // [B_,HID]
    const unsigned short* __restrict__ W,    // [NS,HID]
    const float* __restrict__ bias,          // [NS]
    float* __restrict__ out) {               // [B_,NS]
  const int tid  = threadIdx.x;
  const int lane = tid & 31;
  const int wid  = tid >> 5;
  const int l15  = lane & 15;
  const int hl   = lane >> 4;
  const int tile = blockIdx.x * 8 + wid;      // 0..511
  const int m0 = (tile & 15) * 16;
  const int n0 = (tile >> 4) * 16;

  v8f acc = {};
  const unsigned short* pA = A + (size_t)(m0 + l15) * HID + hl * 8;
  const unsigned short* pB = W + (size_t)(n0 + l15) * HID + hl * 8;

  for (int kc = 0; kc < HID; kc += 32) {
    BFrag a, b;
    a.q[0] = *reinterpret_cast<const U4*>(pA + kc);
    a.q[1] = *reinterpret_cast<const U4*>(pA + kc + 16);
    b.q[0] = *reinterpret_cast<const U4*>(pB + kc);
    b.q[1] = *reinterpret_cast<const U4*>(pB + kc + 16);
    acc = __builtin_amdgcn_wmma_f32_16x16x32_bf16(
        false, a.v, false, b.v, (short)0, acc, false, false);
  }

  const int n = n0 + l15;
  const float bn = bias[n];
#pragma unroll
  for (int r = 0; r < 8; ++r) {
    const int m = m0 + r + 8 * hl;
    out[(size_t)m * NS + n] = acc[r] + bn;
  }
}

// ---------------------------------------------------------------------------
// Host-side orchestration
// ---------------------------------------------------------------------------
extern "C" void kernel_launch(void* const* d_in, const int* in_sizes, int n_in,
                              void* d_out, int out_size, void* d_ws, size_t ws_size,
                              hipStream_t stream) {
  const float* x     = (const float*)d_in[0];   // [B_,NS,TT]
  const float* Wi    = (const float*)d_in[1];   // [HID,NS]
  const float* bi    = (const float*)d_in[2];   // [HID]
  const float* Wh    = (const float*)d_in[3];   // [HID,2*HID]
  const float* bh    = (const float*)d_in[4];   // [HID]
  const float* gamma = (const float*)d_in[5];   // [HID]
  const float* beta  = (const float*)d_in[6];   // [HID]
  const float* Wl    = (const float*)d_in[7];   // [NS,HID]
  const float* bl    = (const float*)d_in[8];   // [NS]
  float* out = (float*)d_out;                   // [B_,NS]

  // Workspace carve-out (all sizes multiples of 256B)
  char* ws = (char*)d_ws;
  size_t off = 0;
  auto carve = [&](size_t bytes) { char* p = ws + off; off += bytes; return p; };
  unsigned short* xt_bf  = (unsigned short*)carve((size_t)TT * B_ * NS * 2);   // 128 MB
  unsigned short* inp_bf = (unsigned short*)carve((size_t)TT * B_ * HID * 2);  // 256 MB
  float*          G      = (float*)carve((size_t)TT * B_ * HID * 4);           // 512 MB
  unsigned short* Wi_bf  = (unsigned short*)carve((size_t)HID * NS * 2);
  unsigned short* Whx_bf = (unsigned short*)carve((size_t)HID * HID * 2);
  unsigned short* Whh_bf = (unsigned short*)carve((size_t)HID * HID * 2);
  unsigned short* Wl_bf  = (unsigned short*)carve((size_t)NS * HID * 2);
  unsigned short* hb0    = (unsigned short*)carve((size_t)B_ * HID * 2);
  unsigned short* hb1    = (unsigned short*)carve((size_t)B_ * HID * 2);
  float*          hF     = (float*)carve((size_t)B_ * HID * 4);
  unsigned short* hn_bf  = (unsigned short*)carve((size_t)B_ * HID * 2);
  float*          meanv  = (float*)carve(HID * 4);
  float*          rstdv  = (float*)carve(HID * 4);
  (void)ws_size; (void)n_in; (void)in_sizes; (void)out_size;

  // 1) Weight conversions to bf16
  cvt_bf16_kernel<<<(HID * NS + 255) / 256, 256, 0, stream>>>(Wi, Wi_bf, HID * NS);
  split_wh_kernel<<<(HID * 2 * HID + 255) / 256, 256, 0, stream>>>(Wh, Whx_bf, Whh_bf);
  cvt_bf16_kernel<<<(NS * HID + 255) / 256, 256, 0, stream>>>(Wl, Wl_bf, NS * HID);

  // 2) Transpose x -> time-major bf16
  {
    dim3 g(TT / 32, NS / 32, B_), b(32, 8);
    transpose_x_kernel<<<g, b, 0, stream>>>(x, xt_bf);
  }

  // 3) inp = X @ Wi^T + bi   (M=T*B, N=HID, K=NS) -> bf16
  {
    dim3 g(HID / 128, (TT * B_) / 128);
    gemm_big_kernel<false, true><<<g, 256, 0, stream>>>(
        xt_bf, Wi_bf, bi, nullptr, inp_bf, TT * B_, HID, NS);
  }

  // 4) G = inp @ Whx^T + bh  (M=T*B, N=HID, K=HID) -> f32
  {
    dim3 g(HID / 128, (TT * B_) / 128);
    gemm_big_kernel<true, false><<<g, 256, 0, stream>>>(
        inp_bf, Whx_bf, bh, G, nullptr, TT * B_, HID, HID);
  }

  // 5) Recurrence: h <- tanh(G_t + h @ Whh^T), ping-pong bf16 h buffers
  zero_bf_kernel<<<(B_ * HID + 255) / 256, 256, 0, stream>>>(hb0, B_ * HID);
  unsigned short* hping[2] = {hb0, hb1};
  for (int t = 0; t < TT; ++t) {
    rnn_step_kernel<<<128, 256, 0, stream>>>(
        G + (size_t)t * B_ * HID, hping[t & 1], Whh_bf, hping[(t + 1) & 1], hF);
  }

  // 6) BatchNorm (batch stats) -> normalized bf16
  bn_stats_kernel<<<(HID + 255) / 256, 256, 0, stream>>>(hF, meanv, rstdv);
  bn_apply_kernel<<<(B_ * HID + 255) / 256, 256, 0, stream>>>(
      hF, meanv, rstdv, gamma, beta, hn_bf);

  // 7) out = hn @ Wl^T + bl
  gemm_out_kernel<<<(B_ / 16) * (NS / 16) / 8, 256, 0, stream>>>(
      hn_bf, Wl_bf, bl, out);
}
